// SparseResNet3D_70729521431109
// MI455X (gfx1250) — compile-verified, matrix operand order checked by reading, not run on hardware
//
#include <hip/hip_runtime.h>
#include <hip/hip_bf16.h>

typedef __attribute__((ext_vector_type(16))) __bf16 v16bf;
typedef __attribute__((ext_vector_type(8)))  float  v8f;

#define WMMA_BF16(A, B, C) \
  __builtin_amdgcn_wmma_f32_16x16x32_bf16(false, (A), false, (B), (short)0, (C), false, false)

__device__ __forceinline__ float lrelu(float v) { return v >= 0.f ? v : 0.01f * v; }

// Direct global -> LDS async copy (16B), tracked by ASYNCcnt. lds_off is the
// wave-relative LDS byte address (low 32 bits of the generic pointer).
__device__ __forceinline__ void async_b128(unsigned lds_off, const void* gaddr) {
  asm volatile("global_load_async_to_lds_b128 %0, %1, off"
               :: "v"(lds_off), "v"(gaddr) : "memory");
}
__device__ __forceinline__ unsigned lds_addr(const void* p) {
  return (unsigned)(uintptr_t)p;  // LDS aperture: low 32 bits = wave-relative offset
}

// Load one 16x32 bf16 fragment (row-major, leading dim ld) per ISA 7.12.2:
// lanes 0-15 rows 0-15 K={0..7,16..23}; lanes 16-31 K={8..15,24..31}.
__device__ __forceinline__ v16bf load_frag(const __bf16* base, int ld, int row,
                                           int kbase, int lane) {
  union { v16bf v; unsigned u[8]; } f;
  int r    = row + (lane & 15);
  int koff = ((lane >> 4) & 1) * 8;
  const __bf16* p = base + r * ld + kbase + koff;
#pragma unroll
  for (int i = 0; i < 4; ++i) {
    f.u[i]     = *(const unsigned*)(p + 2 * i);
    f.u[i + 4] = *(const unsigned*)(p + 16 + 2 * i);
  }
  return f.v;
}

// ---------------------------------------------------------------------------
// Prep: bf16 weight conversion/rearrange + folded BN scale/shift + zero buffer.
// ---------------------------------------------------------------------------
__global__ void prep_kernel(const float* down_w, const float* down_b, const float* dbn,
                            const float* w1, const float* b1, const float* w2,
                            const float* b2, const float* w3, const float* b3,
                            const float* bng, const float* bnb, const float* bnm,
                            const float* bnv, const float* fw,
                            __bf16* wdown, __bf16* w1b, __bf16* w2b, __bf16* w3b,
                            __bf16* wfin, float* dscale, float* dshift,
                            float* bscale, float* bshift, float* zbuf) {
  int tid = threadIdx.x;
  if (tid < 64) zbuf[tid] = 0.f;
  for (int e = tid; e < 64 * 256; e += 256) wdown[e] = (__bf16)down_w[e];
  for (int e = tid; e < 3 * 64 * 64; e += 256) {
    w1b[e] = (__bf16)w1[e];
    w3b[e] = (__bf16)w3[e];
  }
  for (int e = tid; e < 32 * 64; e += 256) wfin[e] = (__bf16)fw[e];
  // [3][64][64][27] -> [3][27][64][64]
  for (int e = tid; e < 3 * 27 * 64 * 64; e += 256) {
    int blk = e / 110592, rem = e % 110592;
    int tap = rem >> 12;
    int o   = (rem >> 6) & 63;
    int i   = rem & 63;
    w2b[e] = (__bf16)w2[((blk * 64 + o) * 64 + i) * 27 + tap];
  }
  for (int c = tid; c < 64; c += 256) {
    float s   = dbn[c] * rsqrtf(dbn[192 + c] + 1e-5f);
    dscale[c] = s;
    dshift[c] = down_b[c] * s + dbn[64 + c] - dbn[128 + c] * s;
  }
  for (int e = tid; e < 3 * 3 * 64; e += 256) {
    int c = e & 63, l = (e >> 6) % 3, blk = e / 192;
    float s = bng[e] * rsqrtf(bnv[e] + 1e-5f);
    const float* bb = (l == 0) ? b1 : (l == 1) ? b2 : b3;
    bscale[e] = s;
    bshift[e] = bb[blk * 64 + c] * s + bnb[e] - bnm[e] * s;
  }
}

// ---------------------------------------------------------------------------
// MaxPool3d(2,2) on mask, threshold > 0.5 -> {0,1} float per 64^3 voxel.
// ---------------------------------------------------------------------------
__global__ void mask_kernel(const float* mask, float* maskf) {
  int v = blockIdx.x * 256 + threadIdx.x;
  int d = v >> 12, h = (v >> 6) & 63, w = v & 63;
  float mmax = -1e30f;
#pragma unroll
  for (int i = 0; i < 2; ++i)
#pragma unroll
    for (int j = 0; j < 2; ++j)
#pragma unroll
      for (int k = 0; k < 2; ++k)
        mmax = fmaxf(mmax, mask[((2 * d + i) * 128 + (2 * h + j)) * 128 + (2 * w + k)]);
  maskf[v] = mmax > 0.5f ? 1.f : 0.f;
}

// ---------------------------------------------------------------------------
// Downsample conv k4 s2 p1 (Cin=4 -> 64), implicit GEMM, K = 256.
// Weight chunks double-buffered via async LDS loads.
// ---------------------------------------------------------------------------
__global__ void down_kernel(const float* feat, const __bf16* wdown,
                            const float* dscale, const float* dshift,
                            const float* maskf, float* xf, __bf16* xmb) {
  __shared__ __bf16 Bl[64 * 256];     // im2col: [voxel][k]  32 KB
  __shared__ __bf16 Al[2][64 * 32];   // weight chunk x2      8 KB
  int rid = blockIdx.x;
  int d = rid >> 6, h = rid & 63;
  int tid = threadIdx.x, wave = tid >> 5, lane = tid & 31;
  int mt = wave & 3, nt0 = (wave >> 2) * 2;

  for (int e = tid; e < 64 * 256; e += 256) {
    int vw = e >> 8, k = e & 255;
    int c = k >> 6, kz = (k >> 4) & 3, ky = (k >> 2) & 3, kx = k & 3;
    int id = 2 * d - 1 + kz, ih = 2 * h - 1 + ky, iw = 2 * vw - 1 + kx;
    float v = 0.f;
    if ((unsigned)id < 128u && (unsigned)ih < 128u && (unsigned)iw < 128u)
      v = feat[((c * 128 + id) * 128 + ih) * 128 + iw];
    Bl[vw * 256 + k] = (__bf16)v;
  }

  int eo = tid * 8, ao_o = eo >> 5, ao_k = eo & 31;  // this thread's A slot
  auto issue_chunk = [&](int c8, int s) {
    async_b128(lds_addr(Al[s] + ao_o * 32 + ao_k), wdown + ao_o * 256 + c8 * 32 + ao_k);
  };
  issue_chunk(0, 0);

  v8f acc0 = {}, acc1 = {};
  for (int c8 = 0; c8 < 8; ++c8) {
    int cur = c8 & 1;
    if (c8 + 1 < 8) {
      issue_chunk(c8 + 1, cur ^ 1);
      asm volatile("s_wait_asynccnt 1" ::: "memory");
    } else {
      asm volatile("s_wait_asynccnt 0" ::: "memory");
    }
    __syncthreads();
    v16bf a  = load_frag(Al[cur], 32, mt * 16, 0, lane);
    v16bf b0 = load_frag(Bl, 256, nt0 * 16, c8 * 32, lane);
    v16bf b1 = load_frag(Bl, 256, (nt0 + 1) * 16, c8 * 32, lane);
    acc0 = WMMA_BF16(a, b0, acc0);
    acc1 = WMMA_BF16(a, b1, acc1);
    __syncthreads();
  }

  int ch0 = mt * 16 + ((lane >> 4) & 1) * 8;
#pragma unroll
  for (int t = 0; t < 2; ++t) {
    v8f acc = t ? acc1 : acc0;
    int vox = rid * 64 + (nt0 + t) * 16 + (lane & 15);
    float mv = maskf[vox];
#pragma unroll
    for (int r = 0; r < 8; ++r) {
      int ch = ch0 + r;
      float v = lrelu(acc[r] * dscale[ch] + dshift[ch]);
      xf[vox * 64 + ch]  = v;
      xmb[vox * 64 + ch] = (__bf16)(v * mv);
    }
  }
}

// ---------------------------------------------------------------------------
// 3x3x3 submanifold conv (64->64), implicit GEMM over 27 taps.
// Double-buffered async LDS staging: tap t+1 streams in while tap t computes.
// ---------------------------------------------------------------------------
__global__ void conv3_kernel(const __bf16* in, const __bf16* Wt,
                             const float* scale, const float* shift,
                             const float* maskf, const float* zerobuf, __bf16* out) {
  __shared__ __bf16 Al[2][64 * 64];   // tap weights x2  16 KB
  __shared__ __bf16 Bl[2][64 * 64];   // shifted acts x2 16 KB
  int rid = blockIdx.x;
  int d = rid >> 6, h = rid & 63;
  int tid = threadIdx.x, wave = tid >> 5, lane = tid & 31;
  int mt = wave & 3, nt0 = (wave >> 2) * 2;
  int vw = tid >> 2, cg = (tid & 3) * 16;  // B staging: voxel vw, channels cg..cg+15

  auto issue_tap = [&](int t, int s) {
    int dz = t / 9 - 1, r9 = t % 9, dy = r9 / 3 - 1, dx = r9 % 3 - 1;
    int id = d + dz, ih = h + dy, iw = vw + dx;
    bool ok = (unsigned)id < 64u && (unsigned)ih < 64u && (unsigned)iw < 64u;
    const __bf16* src = ok ? (in + (((id * 64 + ih) * 64 + iw) * 64 + cg))
                           : (const __bf16*)zerobuf;
    unsigned bo = lds_addr(Bl[s] + vw * 64 + cg);
    async_b128(bo,      src);
    async_b128(bo + 16, ok ? (src + 8) : src);
    const __bf16* wsrc = Wt + t * 4096 + tid * 16;
    unsigned ao = lds_addr(Al[s] + tid * 16);
    async_b128(ao,      wsrc);
    async_b128(ao + 16, wsrc + 8);
  };

  issue_tap(0, 0);
  v8f acc0 = {}, acc1 = {};
  for (int t = 0; t < 27; ++t) {
    int cur = t & 1;
    if (t + 1 < 27) {
      issue_tap(t + 1, cur ^ 1);
      asm volatile("s_wait_asynccnt 4" ::: "memory");  // drain tap t's 4 loads
    } else {
      asm volatile("s_wait_asynccnt 0" ::: "memory");
    }
    __syncthreads();
#pragma unroll
    for (int kc = 0; kc < 64; kc += 32) {
      v16bf a  = load_frag(Al[cur], 64, mt * 16, kc, lane);
      v16bf b0 = load_frag(Bl[cur], 64, nt0 * 16, kc, lane);
      v16bf b1 = load_frag(Bl[cur], 64, (nt0 + 1) * 16, kc, lane);
      acc0 = WMMA_BF16(a, b0, acc0);
      acc1 = WMMA_BF16(a, b1, acc1);
    }
    __syncthreads();
  }

  int ch0 = mt * 16 + ((lane >> 4) & 1) * 8;
#pragma unroll
  for (int t = 0; t < 2; ++t) {
    v8f acc = t ? acc1 : acc0;
    int vox = rid * 64 + (nt0 + t) * 16 + (lane & 15);
    float mv = maskf[vox];
#pragma unroll
    for (int r = 0; r < 8; ++r) {
      int ch = ch0 + r;
      float v = lrelu(acc[r] * scale[ch] + shift[ch]);
      out[vox * 64 + ch] = (__bf16)(v * mv);
    }
  }
}

// ---------------------------------------------------------------------------
// Pointwise GEMM. mode 0: out = bf16(mask*lrelu(BN(W@in)))
//                 mode 1: x = lrelu(x + mask*BN(W@in)); xmb = bf16(x*mask)
//                 mode 2: out_f32[NCDHW] = W@x_f32 + bias   (M = 32)
// ---------------------------------------------------------------------------
__global__ void pw_kernel(const __bf16* inB, const float* inF, const __bf16* W,
                          const float* scale, const float* shift, const float* maskf,
                          float* xf, __bf16* outb, float* outf, int mode) {
  __shared__ __bf16 Al[64 * 64];
  __shared__ __bf16 Bl[64 * 64];
  int rid = blockIdx.x;
  int tid = threadIdx.x, wave = tid >> 5, lane = tid & 31;
  int M = (mode == 2) ? 32 : 64;

  if (mode < 2) {  // bf16 activations: direct async global->LDS
    unsigned bo = lds_addr(Bl + tid * 16);
    const __bf16* src = inB + rid * 4096 + tid * 16;
    async_b128(bo, src);
    async_b128(bo + 16, src + 8);
  } else {  // fp32 -> bf16 conversion during staging
    for (int e = tid; e < 4096; e += 256) Bl[e] = (__bf16)inF[rid * 4096 + e];
  }
  for (int e = tid * 16; e < M * 64; e += 4096) {
    unsigned ao = lds_addr(Al + e);
    async_b128(ao, W + e);
    async_b128(ao + 16, W + e + 8);
  }
  asm volatile("s_wait_asynccnt 0" ::: "memory");
  __syncthreads();

  if (mode < 2) {
    int mt = wave & 3, nt0 = (wave >> 2) * 2;
    v8f acc0 = {}, acc1 = {};
#pragma unroll
    for (int kc = 0; kc < 64; kc += 32) {
      v16bf a  = load_frag(Al, 64, mt * 16, kc, lane);
      v16bf b0 = load_frag(Bl, 64, nt0 * 16, kc, lane);
      v16bf b1 = load_frag(Bl, 64, (nt0 + 1) * 16, kc, lane);
      acc0 = WMMA_BF16(a, b0, acc0);
      acc1 = WMMA_BF16(a, b1, acc1);
    }
    int ch0 = mt * 16 + ((lane >> 4) & 1) * 8;
#pragma unroll
    for (int t = 0; t < 2; ++t) {
      v8f acc = t ? acc1 : acc0;
      int vox = rid * 64 + (nt0 + t) * 16 + (lane & 15);
      float mv = maskf[vox];
#pragma unroll
      for (int r = 0; r < 8; ++r) {
        int ch = ch0 + r;
        float yv = acc[r] * scale[ch] + shift[ch];
        if (mode == 0) {
          outb[vox * 64 + ch] = (__bf16)(lrelu(yv) * mv);
        } else {
          float xn = lrelu(xf[vox * 64 + ch] + mv * yv);
          xf[vox * 64 + ch]   = xn;
          outb[vox * 64 + ch] = (__bf16)(xn * mv);
        }
      }
    }
  } else {  // final projection: 2 M-tiles x 4 N-tiles = 8 tiles, one per wave
    int mt = wave & 1, nt = wave >> 1;
    v8f acc = {};
#pragma unroll
    for (int kc = 0; kc < 64; kc += 32) {
      v16bf a = load_frag(Al, 64, mt * 16, kc, lane);
      v16bf b = load_frag(Bl, 64, nt * 16, kc, lane);
      acc = WMMA_BF16(a, b, acc);
    }
    int vox = rid * 64 + nt * 16 + (lane & 15);
    int ch0 = mt * 16 + ((lane >> 4) & 1) * 8;
#pragma unroll
    for (int r = 0; r < 8; ++r) {
      int ch = ch0 + r;
      outf[ch * 262144 + vox] = acc[r] + shift[ch];
    }
  }
}

// ---------------------------------------------------------------------------
extern "C" void kernel_launch(void* const* d_in, const int* in_sizes, int n_in,
                              void* d_out, int out_size, void* d_ws, size_t ws_size,
                              hipStream_t stream) {
  const float* feat   = (const float*)d_in[0];
  const float* maskin = (const float*)d_in[1];
  const float* down_w = (const float*)d_in[2];
  const float* down_b = (const float*)d_in[3];
  const float* dbn    = (const float*)d_in[4];
  const float* w1     = (const float*)d_in[5];
  const float* b1     = (const float*)d_in[6];
  const float* w2     = (const float*)d_in[7];
  const float* b2     = (const float*)d_in[8];
  const float* w3     = (const float*)d_in[9];
  const float* b3     = (const float*)d_in[10];
  const float* bng    = (const float*)d_in[11];
  const float* bnb    = (const float*)d_in[12];
  const float* bnm    = (const float*)d_in[13];
  const float* bnv    = (const float*)d_in[14];
  const float* fw     = (const float*)d_in[15];
  const float* fb     = (const float*)d_in[16];
  float* out = (float*)d_out;

  size_t off = 0;
  auto carve = [&](size_t bytes) {
    void* p = (char*)d_ws + off;
    off = (off + bytes + 255) & ~(size_t)255;
    return p;
  };
  const size_t NV = 262144;  // 64^3 voxels
  float*  maskf  = (float*)carve(NV * 4);
  float*  xf     = (float*)carve(NV * 64 * 4);
  __bf16* xmb    = (__bf16*)carve(NV * 64 * 2);
  __bf16* y1     = (__bf16*)carve(NV * 64 * 2);
  __bf16* y2     = (__bf16*)carve(NV * 64 * 2);
  __bf16* wdown  = (__bf16*)carve(64 * 256 * 2);
  __bf16* w1b    = (__bf16*)carve(3 * 64 * 64 * 2);
  __bf16* w2b    = (__bf16*)carve(3 * 27 * 64 * 64 * 2);
  __bf16* w3b    = (__bf16*)carve(3 * 64 * 64 * 2);
  __bf16* wfin   = (__bf16*)carve(32 * 64 * 2);
  float*  dscale = (float*)carve(64 * 4);
  float*  dshift = (float*)carve(64 * 4);
  float*  bscale = (float*)carve(3 * 3 * 64 * 4);
  float*  bshift = (float*)carve(3 * 3 * 64 * 4);
  float*  zbuf   = (float*)carve(256);

  prep_kernel<<<1, 256, 0, stream>>>(down_w, down_b, dbn, w1, b1, w2, b2, w3, b3,
                                     bng, bnb, bnm, bnv, fw, wdown, w1b, w2b, w3b,
                                     wfin, dscale, dshift, bscale, bshift, zbuf);
  mask_kernel<<<1024, 256, 0, stream>>>(maskin, maskf);
  down_kernel<<<4096, 256, 0, stream>>>(feat, wdown, dscale, dshift, maskf, xf, xmb);

  for (int b = 0; b < 3; ++b) {
    pw_kernel<<<4096, 256, 0, stream>>>(xmb, nullptr, w1b + b * 4096,
                                        bscale + (b * 3 + 0) * 64, bshift + (b * 3 + 0) * 64,
                                        maskf, nullptr, y1, nullptr, 0);
    conv3_kernel<<<4096, 256, 0, stream>>>(y1, w2b + b * 27 * 4096,
                                           bscale + (b * 3 + 1) * 64, bshift + (b * 3 + 1) * 64,
                                           maskf, zbuf, y2);
    pw_kernel<<<4096, 256, 0, stream>>>(y2, nullptr, w3b + b * 4096,
                                        bscale + (b * 3 + 2) * 64, bshift + (b * 3 + 2) * 64,
                                        maskf, xf, xmb, nullptr, 1);
  }
  // final 1x1: dense fp32 x -> 32 channels, NCDHW output, bias only
  pw_kernel<<<4096, 256, 0, stream>>>(nullptr, xf, wfin, nullptr, fb, maskf,
                                      nullptr, nullptr, out, 2);
}
